// MaskGCNLayer_23802708754536
// MI455X (gfx1250) — matrix operand order repaired and checked by reading, not compiled, not run on hardware
//
#include <hip/hip_runtime.h>
#include <hip/hip_bf16.h>

typedef __attribute__((ext_vector_type(16))) _Float16 v16h;
typedef __attribute__((ext_vector_type(8)))  float    v8f;
typedef __attribute__((ext_vector_type(4)))  unsigned int v4u;
typedef __attribute__((ext_vector_type(8)))  int      v8i;
typedef __attribute__((ext_vector_type(4)))  int      v4i;

#define B_    16
#define N_    1024
#define DOUT_ 64
#define EDIM_ 16
#define WIN_  12

// ---- workspace layout (float units) ----
#define OFF_E        0            // 16384
#define OFF_BIAS_R   16384        // 131072
#define OFF_BIAS_U   147456       // 65536
#define OFF_SUPPORTS 212992       // 1048576
#define OFF_EM       1261568      // 1048576
#define OFF_XIN      2310144      // 2097152
#define OFF_XB       4407296      // 2097152
#define OFF_ZR       6504448      // 2097152
#define OFF_CAND     8601600      // 2097152
#define OFF_CNT      10698752     // 32 u32 counters (16 mask counts + 16 removal counts)

// ---- output layout (float units): h, mask, mask_loss ----
#define OUT_MASK 1048576
#define OUT_LOSS 17825792

// ======================= reductions (wave32) =======================
__device__ __forceinline__ float waveMax(float v) {
#pragma unroll
  for (int o = 16; o > 0; o >>= 1) v = fmaxf(v, __shfl_xor(v, o, 32));
  return v;
}
__device__ __forceinline__ float waveSum(float v) {
#pragma unroll
  for (int o = 16; o > 0; o >>= 1) v += __shfl_xor(v, o, 32);
  return v;
}
__device__ __forceinline__ float blockRedMax(float v, float* red) {
  int lane = threadIdx.x & 31, wid = threadIdx.x >> 5;
  v = waveMax(v);
  if (lane == 0) red[wid] = v;
  __syncthreads();
  float r = red[0];
#pragma unroll
  for (int i = 1; i < 8; ++i) r = fmaxf(r, red[i]);
  __syncthreads();
  return r;
}
__device__ __forceinline__ float blockRedSum(float v, float* red) {
  int lane = threadIdx.x & 31, wid = threadIdx.x >> 5;
  v = waveSum(v);
  if (lane == 0) red[wid] = v;
  __syncthreads();
  float r = 0.f;
#pragma unroll
  for (int i = 0; i < 8; ++i) r += red[i];
  __syncthreads();
  return r;
}

// ======================= threefry2x32 (JAX-exact) =======================
__device__ __forceinline__ unsigned rotl32(unsigned x, int r) { return (x << r) | (x >> (32 - r)); }
__device__ __forceinline__ void threefry2x32(unsigned k0, unsigned k1, unsigned x0, unsigned x1,
                                             unsigned* y0, unsigned* y1) {
  unsigned ks0 = k0, ks1 = k1, ks2 = k0 ^ k1 ^ 0x1BD11BDAu;
  x0 += ks0; x1 += ks1;
#define TF_R(a,b,c,d) { x0+=x1; x1=rotl32(x1,a); x1^=x0; x0+=x1; x1=rotl32(x1,b); x1^=x0; \
                        x0+=x1; x1=rotl32(x1,c); x1^=x0; x0+=x1; x1=rotl32(x1,d); x1^=x0; }
  TF_R(13,15,26,6);  x0 += ks1; x1 += ks2 + 1u;
  TF_R(17,29,16,24); x0 += ks2; x1 += ks0 + 2u;
  TF_R(13,15,26,6);  x0 += ks0; x1 += ks1 + 3u;
  TF_R(17,29,16,24); x0 += ks1; x1 += ks2 + 4u;
  TF_R(13,15,26,6);  x0 += ks2; x1 += ks0 + 5u;
#undef TF_R
  *y0 = x0; *y1 = x1;
}
__device__ __forceinline__ float bits_to_uniform(unsigned b) {
  return __uint_as_float(0x3f800000u | (b >> 9)) - 1.0f;
}

// ======================= prep: E = we + pos_encoding; zero counters ===========
__global__ void k_prep(const float* __restrict__ we, const int* __restrict__ itp,
                       float* __restrict__ ws) {
  int idx = blockIdx.x * 256 + threadIdx.x;
  if (blockIdx.x == 0 && threadIdx.x < 32)
    ((unsigned*)(ws + OFF_CNT))[threadIdx.x] = 0u;
  if (idx < N_ * EDIM_) {
    int d = idx & (EDIM_ - 1);
    float pos = (float)itp[0];
    int j = d >> 1;
    float div = expf((float)(2 * j) * (-logf(10000.0f) / (float)EDIM_));
    float pe = (d & 1) ? cosf(pos * div) : sinf(pos * div);
    ws[OFF_E + idx] = we[idx] + pe;
  }
}

// ======================= biases: E @ b_pool_{r,u} =======================
__global__ void k_bias(const float* __restrict__ bpr, const float* __restrict__ bpu,
                       float* __restrict__ ws) {
  int idx = blockIdx.x * 256 + threadIdx.x;
  if (idx >= N_ * 192) return;
  int n = idx / 192, r = idx % 192;
  float acc = 0.f;
  if (r < 128) {
#pragma unroll
    for (int d = 0; d < EDIM_; ++d) acc += ws[OFF_E + n * 16 + d] * bpr[d * 128 + r];
    ws[OFF_BIAS_R + n * 128 + r] = acc;
  } else {
    int o = r - 128;
#pragma unroll
    for (int d = 0; d < EDIM_; ++d) acc += ws[OFF_E + n * 16 + d] * bpu[d * 64 + o];
    ws[OFF_BIAS_U + n * 64 + o] = acc;
  }
}

// ======================= xin = concat(x, state) =======================
__global__ void k_xin(const float* __restrict__ x, const float* __restrict__ st,
                      float* __restrict__ ws) {
  int idx = blockIdx.x * 256 + threadIdx.x;  // B*N*128
  int row = idx >> 7, i = idx & 127;
  ws[OFF_XIN + idx] = (i < 64) ? x[row * 64 + i] : st[row * 64 + (i - 64)];
}

// ======================= cand = concat(x, z*state) =======================
__global__ void k_cand(const float* __restrict__ x, const float* __restrict__ st,
                       float* __restrict__ ws) {
  int idx = blockIdx.x * 256 + threadIdx.x;
  int row = idx >> 7, i = idx & 127;
  float v;
  if (i < 64) v = x[row * 64 + i];
  else {
    float z = ws[OFF_ZR + row * 128 + (i - 64)];
    v = z * st[row * 64 + (i - 64)];
  }
  ws[OFF_CAND + idx] = v;
}

// ======== supports = softmax(relu(E E^T)) rows; em = E @ mask_pool ========
__global__ __launch_bounds__(256) void k_supports(const float* __restrict__ mask_pool,
                                                  float* __restrict__ ws) {
  __shared__ float srow[1024];
  __shared__ float red[8];
  int n = blockIdx.x, tid = threadIdx.x;
  float en[EDIM_];
#pragma unroll
  for (int d = 0; d < EDIM_; ++d) en[d] = ws[OFF_E + n * 16 + d];
  float lmax = -1e30f;
  for (int m = tid; m < 1024; m += 256) {
    float s = 0.f, e = 0.f;
#pragma unroll
    for (int d = 0; d < EDIM_; ++d) {
      s += en[d] * ws[OFF_E + m * 16 + d];
      e += en[d] * mask_pool[d * 1024 + m];
    }
    ws[OFF_EM + n * 1024 + m] = e;
    float rg = fmaxf(s, 0.f);
    srow[m] = rg;
    lmax = fmaxf(lmax, rg);
  }
  float mx = blockRedMax(lmax, red);
  float ls = 0.f;
  for (int m = tid; m < 1024; m += 256) {
    float e = expf(srow[m] - mx);
    srow[m] = e; ls += e;
  }
  float sm = blockRedSum(ls, red);
  float inv = 1.f / sm;
  for (int m = tid; m < 1024; m += 256) ws[OFF_SUPPORTS + n * 1024 + m] = srow[m] * inv;
}

// ==== fused: mask + time-softmax + sup row-tile (LDS) + sup@xin via WMMA ====
// MODE 0: compute+write mask, count actives, xin = concat(x,state) -> xb_r
// MODE 1: read mask,                       xin = cand             -> xb_u
template <int MODE>
__global__ __launch_bounds__(256) void k_fused(const float* __restrict__ Tg,
                                               const float* __restrict__ time_pool,
                                               const float* __restrict__ bn_gamma,
                                               const float* __restrict__ bn_beta,
                                               const float* __restrict__ ws,
                                               const float* __restrict__ xin,
                                               float* __restrict__ xb,
                                               float* __restrict__ mask_out,
                                               unsigned* __restrict__ cnt) {
  extern __shared__ float smem[];
  const int SUPS = 1028;                    // padded stride: conflict-free A-frag reads
  float* Tl  = smem;                        // 12*1024 (LDS offset 0 -> TDM target)
  float* sup = smem + 12 * 1024;            // 16*1028
  float* red = sup + 16 * SUPS;             // 8
  int b  = blockIdx.x >> 6;
  int n0 = (blockIdx.x & 63) << 4;
  int tid = threadIdx.x, lane = tid & 31, wid = tid >> 5;

  // ---- stage T[b] (12x1024 f32 = 48KB) into LDS via the Tensor Data Mover ----
#if __has_builtin(__builtin_amdgcn_tensor_load_to_lds)
  if (wid == 0) {
    unsigned long long ga = (unsigned long long)(Tg + b * 12 * 1024);
    // D# group 0: count=1 | lds_addr=0 | global_addr | type=2
    v4u g0 = {1u, 0u, (unsigned)ga,
              (unsigned)((ga >> 32) & 0x01FFFFFFu) | (2u << 30)};
    // D# group 1: data_size=4B; tensor 1024x12; tile 1024x12; strides 1024 / 12288
    v8i g1 = {(int)(2u << 16),        // data_size = 2 (4 bytes), wg_mask=0
              (int)(1024u << 16),     // tensor_dim0[15:0]
              (int)(12u << 16),       // tensor_dim0 hi=0 | tensor_dim1[15:0]=12
              (int)(1024u << 16),     // tensor_dim1 hi=0 | tile_dim0=1024
              12,                     // tile_dim1=12 | tile_dim2=0
              1024,                   // tensor_dim0_stride[31:0]
              (int)(12288u << 16),    // dim0_stride hi=0 | tensor_dim1_stride[15:0]
              0};
    v4i gz = {0, 0, 0, 0};
#if defined(__clang_major__) && __clang_major__ >= 23
    v8i gz8 = {0, 0, 0, 0, 0, 0, 0, 0};
    __builtin_amdgcn_tensor_load_to_lds(g0, g1, gz, gz, gz8, 0);
#else
    __builtin_amdgcn_tensor_load_to_lds(g0, g1, gz, gz, 0);
#endif
    __builtin_amdgcn_s_wait_tensorcnt(0);
  }
  __syncthreads();
#else
  for (int i = tid; i < 12 * 1024; i += 256) Tl[i] = Tg[b * 12 * 1024 + i];
  __syncthreads();
#endif

  const float invs = 0.9999950000374997f;   // 1/sqrt(1+1e-5)
  int lcnt = 0;
  for (int r = 0; r < 16; ++r) {
    int n = n0 + r;
    float tn[WIN_];
#pragma unroll
    for (int w = 0; w < WIN_; ++w) tn[w] = Tl[w * 1024 + n];
    float g_n = bn_gamma[n], be_n = bn_beta[n];
    float lmax = -1e30f;
    for (int m = tid; m < 1024; m += 256) {
      float g = 0.f;
#pragma unroll
      for (int w = 0; w < WIN_; ++w) g += tn[w] * Tl[w * 1024 + m];
      float rg = fmaxf(g, 0.f);
      sup[r * SUPS + m] = rg;
      lmax = fmaxf(lmax, rg);
      if (MODE == 0) {
        float tm = 0.f;
#pragma unroll
        for (int w = 0; w < WIN_; ++w) tm += tn[w] * time_pool[w * 1024 + m];
        float raw = ws[OFF_EM + n * 1024 + m] + tm;
        raw = raw * invs * g_n + be_n;
        float s  = 1.f / (1.f + expf(-(raw + 3.0f)));
        float sc = s * 1.2f - 0.2f;
        float mv = rintf(fminf(fmaxf(sc, 0.f), 1.f));
        mask_out[(b * 1024 + n) * 1024 + m] = mv;
        lcnt += (int)mv;
      }
    }
    float rowmax = blockRedMax(lmax, red);
    float lsum = 0.f;
    for (int m = tid; m < 1024; m += 256) {
      float e = expf(sup[r * SUPS + m] - rowmax);
      sup[r * SUPS + m] = e; lsum += e;
    }
    float inv = 1.f / blockRedSum(lsum, red);
    for (int m = tid; m < 1024; m += 256) {
      float mv = mask_out[(b * 1024 + n) * 1024 + m];  // same thread wrote it (MODE 0)
      sup[r * SUPS + m] = 0.5f * (ws[OFF_SUPPORTS + n * 1024 + m] + sup[r * SUPS + m] * inv) * mv;
    }
  }
  if (MODE == 0) {
    float c = blockRedSum((float)lcnt, red);
    if (tid == 0) atomicAdd(&cnt[b], (unsigned)(c + 0.5f));
  }
  __syncthreads();

  // ---- GEMM: xb[16,128] = sup[16,1024] @ xin_b[1024,128], one 16x16 tile/wave ----
  int col0 = wid * 16;
  int arow = lane & 15;
  int koff = (lane < 16) ? 0 : 8;
  int kb   = (lane < 16) ? 0 : 16;
  int bcol = col0 + (lane & 15);
  const float* xinB = xin + b * 1024 * 128;
  v8f c = {};
  for (int k0 = 0; k0 < 1024; k0 += 32) {
    v16h a, bf;
#pragma unroll
    for (int i = 0; i < 8; ++i) {
      a[i]     = (_Float16)sup[arow * SUPS + k0 + koff + i];
      a[i + 8] = (_Float16)sup[arow * SUPS + k0 + 16 + koff + i];
    }
#pragma unroll
    for (int i = 0; i < 16; ++i)
      bf[i] = (_Float16)xinB[(k0 + (i < 8 ? koff + i : 8 + koff + i)) * 128 + bcol];
    c = __builtin_amdgcn_wmma_f32_16x16x32_f16(false, a, false, bf, (short)0, c, false, false);
  }
  int crow = 8 * (lane >> 4);
#pragma unroll
  for (int v = 0; v < 8; ++v)
    xb[(b * 1024 + n0 + crow + v) * 128 + bcol] = c[v];
}

// ==== gate GEMM: out[b,n,o] = act( sum_d (E[n,d]*xg[row,:]) @ Wpool[d] + bias ) ====
// MODE 0: O=128, sigmoid -> zr.   MODE 1: O=64, h = r*state + (1-r)*tanh(.) -> out
template <int MODE>
__global__ __launch_bounds__(256) void k_gate(const float* __restrict__ W4,
                                              const float* __restrict__ biasv,
                                              const float* __restrict__ Ew,
                                              const float* __restrict__ xg_a,
                                              const float* __restrict__ xbv,
                                              float* __restrict__ zr,
                                              const float* __restrict__ state,
                                              float* __restrict__ outh) {
  const int O = (MODE == 0) ? 128 : 64;
  extern __shared__ float smem[];
  const int AS = 258;                 // padded row stride
  float* Axg = smem;                  // 64 x 258
  float* Er  = smem + 64 * AS;        // 64 x 16
  int tid = threadIdx.x, lane = tid & 31, wid = tid >> 5;
  int row0 = blockIdx.x * 64;

  for (int idx = tid; idx < 64 * 256; idx += 256) {
    int r = idx >> 8, ki = idx & 255;
    int row = row0 + r;
    Axg[r * AS + ki] = (ki < 128) ? xg_a[row * 128 + ki] : xbv[row * 128 + (ki - 128)];
  }
  for (int idx = tid; idx < 64 * 16; idx += 256) {
    int r = idx >> 4, d = idx & 15;
    Er[idx] = Ew[OFF_E + (((row0 + r) & 1023) << 4) + d];
  }
  __syncthreads();

  const int NCT = O / 16;            // col tiles: 8 (r) or 4 (u)
  const int WPC = 8 / NCT;           // waves per col tile
  const int MT  = 4 / WPC;           // 16-row m-tiles per wave (rows = 64)
  int colTile = wid % NCT;
  int rhalf   = wid / NCT;
  int col  = colTile * 16 + (lane & 15);
  int koff = (lane < 16) ? 0 : 8;
  int kb   = (lane < 16) ? 0 : 16;
  v8f acc[4];
#pragma unroll
  for (int t = 0; t < MT; ++t) acc[t] = v8f{};

  for (int k4 = 0; k4 < 4096; k4 += 32) {
    int d = k4 >> 8;
    v16h bf;
#pragma unroll
    for (int i = 0; i < 16; ++i)
      bf[i] = (_Float16)W4[(k4 + (i < 8 ? koff + i : 8 + koff + i)) * O + col];
    if (k4 + 32 < 4096) __builtin_prefetch(&W4[(k4 + 32 + kb) * O + col], 0, 1);
#pragma unroll
    for (int t = 0; t < MT; ++t) {
      int rl = (rhalf * MT + t) * 16 + (lane & 15);
      float e = Er[rl * 16 + d];
      v16h a;
#pragma unroll
      for (int i = 0; i < 8; ++i) {
        int ki1 = (k4 + koff + i) & 255;
        int ki2 = (k4 + 16 + koff + i) & 255;
        a[i]     = (_Float16)(e * Axg[rl * AS + ki1]);
        a[i + 8] = (_Float16)(e * Axg[rl * AS + ki2]);
      }
      acc[t] = __builtin_amdgcn_wmma_f32_16x16x32_f16(false, a, false, bf, (short)0, acc[t],
                                                      false, false);
    }
  }

  int crow = 8 * (lane >> 4);
#pragma unroll
  for (int t = 0; t < MT; ++t) {
    int rbase = (rhalf * MT + t) * 16 + crow;
#pragma unroll
    for (int v = 0; v < 8; ++v) {
      int rg = row0 + rbase + v;
      int n = rg & 1023;
      float val = acc[t][v] + biasv[n * O + col];
      if (MODE == 0) {
        zr[rg * 128 + col] = 1.f / (1.f + expf(-val));
      } else {
        float hc = tanhf(val);
        float rr = zr[rg * 128 + 64 + col];
        outh[rg * 64 + col] = rr * state[rg * 64 + col] + (1.f - rr) * hc;
      }
    }
  }
}

// ======================= removal counting (threefry, exact JAX key 42) =========
__global__ void k_removal(const float* __restrict__ mask, float* __restrict__ ws) {
  __shared__ unsigned sc[16];
  __shared__ float sprob;
  unsigned* cnt = (unsigned*)(ws + OFF_CNT);
  unsigned* rem = cnt + 16;
  if (threadIdx.x < 16) sc[threadIdx.x] = 0u;
  if (threadIdx.x == 0) {
    unsigned tot = 0;
#pragma unroll
    for (int i = 0; i < 16; ++i) tot += cnt[i];
    float an = (float)tot / 16777216.0f;
    sprob = fminf(0.05f / an + 1e-8f, 1.0f);
  }
  __syncthreads();
  unsigned j = blockIdx.x * 256 + threadIdx.x;       // [0, 2^23)
  unsigned y0, y1;
  threefry2x32(0u, 42u, j, j + 8388608u, &y0, &y1);
  float u0 = bits_to_uniform(y0), u1 = bits_to_uniform(y1);
  unsigned p0 = j, p1 = j + 8388608u;
  if (mask[p0] > 0.f && u0 < sprob) atomicAdd(&sc[p0 >> 20], 1u);
  if (mask[p1] > 0.f && u1 < sprob) atomicAdd(&sc[p1 >> 20], 1u);
  __syncthreads();
  if (threadIdx.x < 16 && sc[threadIdx.x]) atomicAdd(&rem[threadIdx.x], sc[threadIdx.x]);
}

// ======================= mask_loss =======================
__global__ void k_loss(const float* __restrict__ ws, float* __restrict__ out_loss) {
  const unsigned* cnt = (const unsigned*)(ws + OFF_CNT);
  const unsigned* rem = cnt + 16;
  if (threadIdx.x < 16) {
    unsigned tot = 0;
#pragma unroll
    for (int i = 0; i < 16; ++i) tot += cnt[i];
    float an = (float)tot / 16777216.0f;
    float extra = fmaxf(0.05f / an - 1.0f, 0.0f);
    unsigned c = cnt[threadIdx.x], r = rem[threadIdx.x];
    out_loss[threadIdx.x] = (float)(c - r) / ((float)c + 1e-8f) + extra;
  }
}

// ======================= launcher =======================
extern "C" void kernel_launch(void* const* d_in, const int* in_sizes, int n_in,
                              void* d_out, int out_size, void* d_ws, size_t ws_size,
                              hipStream_t stream) {
  (void)in_sizes; (void)n_in; (void)out_size; (void)ws_size;
  const float* x      = (const float*)d_in[0];
  const float* state  = (const float*)d_in[1];
  const float* we     = (const float*)d_in[2];
  const float* T      = (const float*)d_in[3];
  const float* Wpr    = (const float*)d_in[4];   // [16,2,128,128] -> [4096,128]
  const float* bpr    = (const float*)d_in[5];
  const float* Wpu    = (const float*)d_in[6];   // [16,2,128,64]  -> [4096,64]
  const float* bpu    = (const float*)d_in[7];
  const float* mpool  = (const float*)d_in[8];
  const float* tpool  = (const float*)d_in[9];
  const float* gamma  = (const float*)d_in[10];
  const float* beta   = (const float*)d_in[11];
  const int*   iter   = (const int*)d_in[12];

  float* ws  = (float*)d_ws;
  float* out = (float*)d_out;
  float* out_h    = out;
  float* out_mask = out + OUT_MASK;
  float* out_loss = out + OUT_LOSS;
  unsigned* cnt = (unsigned*)(ws + OFF_CNT);

  const size_t DSH = (size_t)(12 * 1024 + 16 * 1028 + 16) * sizeof(float);  // ~115 KB
  const size_t FSH = (size_t)(64 * 258 + 64 * 16) * sizeof(float);          // ~70 KB

  k_prep<<<64, 256, 0, stream>>>(we, iter, ws);
  k_bias<<<(N_ * 192 + 255) / 256, 256, 0, stream>>>(bpr, bpu, ws);
  k_xin<<<8192, 256, 0, stream>>>(x, state, ws);
  k_supports<<<1024, 256, 0, stream>>>(mpool, ws);

  // mask + sup tile + xb_r = sup @ concat(x,state)
  k_fused<0><<<1024, 256, DSH, stream>>>(T, tpool, gamma, beta, ws,
                                         ws + OFF_XIN, ws + OFF_XB, out_mask, cnt);
  // mask statistics (need counts from k_fused<0>)
  k_removal<<<32768, 256, 0, stream>>>(out_mask, ws);
  k_loss<<<1, 32, 0, stream>>>(ws, out_loss);

  // r-gate: zr = sigmoid(gate(concat(x,state), xb_r))
  k_gate<0><<<256, 256, FSH, stream>>>(Wpr, ws + OFF_BIAS_R, ws,
                                       ws + OFF_XIN, ws + OFF_XB,
                                       ws + OFF_ZR, state, out_h);
  // candidate = concat(x, z*state)
  k_cand<<<8192, 256, 0, stream>>>(x, state, ws);
  // xb_u = sup @ cand (mask re-read from output)
  k_fused<1><<<1024, 256, DSH, stream>>>(T, tpool, gamma, beta, ws,
                                         ws + OFF_CAND, ws + OFF_XB, out_mask, cnt);
  // u-gate + GRU blend epilogue -> h
  k_gate<1><<<256, 256, FSH, stream>>>(Wpu, ws + OFF_BIAS_U, ws,
                                       ws + OFF_CAND, ws + OFF_XB,
                                       ws + OFF_ZR, state, out_h);
}
